// SubtasksRecurrence_22823456211362
// MI455X (gfx1250) — compile-verified, matrix operand order checked by reading, not run on hardware
//
#include <hip/hip_runtime.h>

// ---------------- types ----------------
typedef __bf16 v16bf __attribute__((ext_vector_type(16)));
typedef float  v8f   __attribute__((ext_vector_type(8)));

union V16U { v16bf v; uint4 q[2]; };

#define ROTL32(x,n) (((x)<<(n))|((x)>>(32-(n))))

// Problem constants
#define TT   128
#define BB   256
#define HH_  256
#define CONVK 4096
#define NS   16
#define SS   32
#define TOT  338     // 16+32+256+32+1+1
#define NPAD 272     // 17 tiles of 16: 256 W_f cols + 2 W_beta cols + 14 pad
#define NT1  17

// ---------------- threefry2x32 (JAX construction) ----------------
__device__ __forceinline__ uint2 threefry2x32(uint2 k, uint2 c) {
  unsigned ks0 = k.x, ks1 = k.y, ks2 = k.x ^ k.y ^ 0x1BD11BDAu;
  unsigned x0 = c.x + ks0, x1 = c.y + ks1;
  const int ra0=13, ra1=15, ra2=26, ra3=6;
  const int rb0=17, rb1=29, rb2=16, rb3=24;
#define TF_R(r) { x0 += x1; x1 = ROTL32(x1, r); x1 ^= x0; }
  TF_R(ra0) TF_R(ra1) TF_R(ra2) TF_R(ra3)  x0 += ks1; x1 += ks2 + 1u;
  TF_R(rb0) TF_R(rb1) TF_R(rb2) TF_R(rb3)  x0 += ks2; x1 += ks0 + 2u;
  TF_R(ra0) TF_R(ra1) TF_R(ra2) TF_R(ra3)  x0 += ks0; x1 += ks1 + 3u;
  TF_R(rb0) TF_R(rb1) TF_R(rb2) TF_R(rb3)  x0 += ks1; x1 += ks2 + 4u;
  TF_R(ra0) TF_R(ra1) TF_R(ra2) TF_R(ra3)  x0 += ks2; x1 += ks0 + 5u;
#undef TF_R
  return make_uint2(x0, x1);
}

__device__ __forceinline__ unsigned tf_bits(uint2 key, unsigned idx, unsigned half) {
  if (idx < half) { return threefry2x32(key, make_uint2(idx, idx + half)).x; }
  else            { return threefry2x32(key, make_uint2(idx - half, idx)).y; }
}

__device__ __forceinline__ float tf_uniform(unsigned bits) {
  return __uint_as_float((bits >> 9) | 0x3f800000u) - 1.0f;
}

__device__ __forceinline__ float sigmoidf_(float x) { return 1.0f / (1.0f + __expf(-x)); }

// ---------------- prep: keys = split(key(42), 256) ----------------
__global__ void k_keys(unsigned* __restrict__ flat) {
  int i = threadIdx.x;                    // 256 threads
  uint2 o = threefry2x32(make_uint2(0u, 42u),
                         make_uint2((unsigned)i, (unsigned)(i + 256)));
  flat[i] = o.x; flat[256 + i] = o.y;     // key_j = (flat[2j], flat[2j+1])
}

// ---------------- prep: pack [W_f_x | W_beta_x | 0] into bf16 WMMA-B tiles ----
// layout: [kt(128)][nt(17)][512]
__global__ void k_pack_wf(const float* __restrict__ Wf, const float* __restrict__ Wbeta,
                          __bf16* __restrict__ wp) {
  size_t e = (size_t)blockIdx.x * 256 + threadIdx.x;
  if (e >= (size_t)128 * NT1 * 512) return;
  int kt   = (int)(e / (NT1 * 512));
  int rem  = (int)(e % (NT1 * 512));
  int nt   = rem / 512;
  int e512 = rem % 512;
  int lane = e512 >> 4, el = e512 & 15;
  int n = nt * 16 + (lane & 15);
  int k = kt * 32 + ((lane < 16) ? 0 : 8) + ((el < 8) ? el : el + 8);
  float v = 0.0f;
  if      (n < 256) v = Wf[(size_t)n * 4161 + k];
  else if (n < 258) v = Wbeta[(size_t)(n - 256) * 4128 + k];
  wp[e] = (__bf16)v;
}

// ---------------- prep: pack a (rows x cols) fp32 weight into bf16 B tiles ----
// layout: [nt(rows/16)][kt(cols/32)][512]
__global__ void k_pack_w(const float* __restrict__ W, __bf16* __restrict__ wp,
                         int rows, int cols) {
  size_t e = (size_t)blockIdx.x * 256 + threadIdx.x;
  if (e >= (size_t)rows * cols) return;
  int ktTiles = cols / 32;
  int nt   = (int)(e / ((size_t)ktTiles * 512));
  int rem  = (int)(e % ((size_t)ktTiles * 512));
  int kt   = rem / 512;
  int e512 = rem % 512;
  int lane = e512 >> 4, el = e512 & 15;
  int n = nt * 16 + (lane & 15);
  int k = kt * 32 + ((lane < 16) ? 0 : 8) + ((el < 8) ? el : el + 8);
  wp[e] = (__bf16)W[(size_t)n * cols + k];
}

// ---- async stage: copy one 17-tile weight slab (17408 B) global -> LDS ------
// Uses CDNA5 GLOBAL_LOAD_ASYNC_TO_LDS_B128 (ASYNCcnt-tracked, no VGPR bounce).
__device__ __forceinline__ void async_stage(const __bf16* __restrict__ src,
                                            __bf16* dstLds, int tid) {
  unsigned ldsbase = (unsigned)(unsigned long long)dstLds;  // LDS byte offset
  unsigned long long g = (unsigned long long)src;
#pragma unroll
  for (int i = 0; i < 5; i++) {
    int idx = tid + i * 256;
    if (idx < (NT1 * 512 * 2) / 16) {
      unsigned loff = ldsbase + (unsigned)idx * 16u;
      unsigned long long ga = g + (unsigned long long)idx * 16ull;
      asm volatile("global_load_async_to_lds_b128 %0, %1, off"
                   :: "v"(loff), "v"(ga) : "memory");
    }
  }
}
__device__ __forceinline__ void wait_async0() {
  asm volatile("s_wait_asynccnt 0x0" ::: "memory");
}

// ---------------- phase 1: SX = obs(32768x4096) @ Wpack(4096x272) ------------
// 256 blocks x 256 threads (8 waves). Each wave owns a 16-row strip, all 17
// N-tiles in f32 accumulators. Weight K-slab is async-double-buffered in LDS.
__global__ __launch_bounds__(256) void k_sx_gemm(const float* __restrict__ obs,
                                                 const __bf16* __restrict__ wpack,
                                                 float* __restrict__ sx) {
  __shared__ __align__(16) __bf16 lb[2][NT1 * 512];   // 2 x 17 KB
  const int tid = threadIdx.x, lane = tid & 31, wv = tid >> 5;
  const size_t rowStrip = (size_t)blockIdx.x * 128 + (size_t)wv * 16;
  const int arow = lane & 15;
  const int koff = (lane < 16) ? 0 : 8;
  const float* aBase = obs + (rowStrip + arow) * CONVK;

  v8f acc[NT1];
#pragma unroll
  for (int i = 0; i < NT1; i++) {
#pragma unroll
    for (int j = 0; j < 8; j++) acc[i][j] = 0.0f;
  }

  // prologue: stage slab 0
  async_stage(wpack, lb[0], tid);
  wait_async0();
  __syncthreads();

  for (int kt = 0; kt < CONVK / 32; ++kt) {
    const __bf16* cur = lb[kt & 1];
    // kick off async load of next slab into the other buffer (overlaps compute)
    if (kt + 1 < CONVK / 32)
      async_stage(wpack + (size_t)(kt + 1) * NT1 * 512, lb[(kt + 1) & 1], tid);

    // A fragment: fp32 -> bf16 on the fly, documented 16-bit A layout
    const float4* p0 = (const float4*)(aBase + kt * 32 + koff);
    float4 f0 = p0[0], f1 = p0[1];
    const float4* p1 = (const float4*)(aBase + kt * 32 + koff + 16);
    float4 f2 = p1[0], f3 = p1[1];
    V16U a;
    a.v[0]=(__bf16)f0.x; a.v[1]=(__bf16)f0.y; a.v[2]=(__bf16)f0.z; a.v[3]=(__bf16)f0.w;
    a.v[4]=(__bf16)f1.x; a.v[5]=(__bf16)f1.y; a.v[6]=(__bf16)f1.z; a.v[7]=(__bf16)f1.w;
    a.v[8]=(__bf16)f2.x; a.v[9]=(__bf16)f2.y; a.v[10]=(__bf16)f2.z; a.v[11]=(__bf16)f2.w;
    a.v[12]=(__bf16)f3.x; a.v[13]=(__bf16)f3.y; a.v[14]=(__bf16)f3.z; a.v[15]=(__bf16)f3.w;

    // pipelined 17 WMMAs: load B tile nt+1 before issuing WMMA nt
    V16U bA, bB;
    {
      const __bf16* bp = cur + lane * 16;
      bA.q[0] = *(const uint4*)bp;
      bA.q[1] = *(const uint4*)(bp + 8);
    }
#pragma unroll
    for (int nt = 0; nt < NT1; nt++) {
      V16U& curb = (nt & 1) ? bB : bA;
      V16U& nxtb = (nt & 1) ? bA : bB;
      if (nt + 1 < NT1) {
        const __bf16* bp = cur + (nt + 1) * 512 + lane * 16;
        nxtb.q[0] = *(const uint4*)bp;
        nxtb.q[1] = *(const uint4*)(bp + 8);
      }
      acc[nt] = __builtin_amdgcn_wmma_f32_16x16x32_bf16(
          false, a.v, false, curb.v, (short)0, acc[nt], false, false);
    }

    wait_async0();       // next slab resident in LDS
    __syncthreads();     // all waves done reading cur / writing next
  }

  // epilogue: C/D layout -> row-major SX
#pragma unroll
  for (int nt = 0; nt < NT1; nt++) {
#pragma unroll
    for (int v = 0; v < 8; v++) {
      size_t row = rowStrip + (size_t)((lane < 16) ? v : v + 8);
      sx[row * NPAD + nt * 16 + (lane & 15)] = acc[nt][v];
    }
  }
}

// ---------------- phase 2 helper: 16x768 = A(16x256,bf16) @ Wpack^T ----------
__device__ __forceinline__ void gru_gemm(const __bf16* __restrict__ Abf,
                                         const __bf16* __restrict__ Wp,
                                         float* __restrict__ Out,
                                         int lane, int wv) {
  const int arow = lane & 15;
  const int koff = (lane < 16) ? 0 : 8;
  V16U afr[8];
#pragma unroll
  for (int kt = 0; kt < 8; kt++) {
    const __bf16* rp = Abf + arow * 256 + kt * 32 + koff;
    afr[kt].q[0] = *(const uint4*)rp;
    afr[kt].q[1] = *(const uint4*)(rp + 16);
  }
  for (int i = 0; i < 6; i++) {          // 48 N-tiles / 8 waves
    int nt = wv * 6 + i;
    v8f acc;
#pragma unroll
    for (int j = 0; j < 8; j++) acc[j] = 0.0f;
    const __bf16* wb = Wp + (size_t)nt * 8 * 512 + lane * 16;
#pragma unroll
    for (int kt = 0; kt < 8; kt++) {
      V16U b;
      b.q[0] = *(const uint4*)(wb + (size_t)kt * 512);
      b.q[1] = *(const uint4*)(wb + (size_t)kt * 512 + 8);
      acc = __builtin_amdgcn_wmma_f32_16x16x32_bf16(
          false, afr[kt].v, false, b.v, (short)0, acc, false, false);
    }
#pragma unroll
    for (int v = 0; v < 8; v++) {
      int r_ = (lane < 16) ? v : v + 8;
      Out[r_ * 768 + nt * 16 + (lane & 15)] = acc[v];
    }
  }
}

// ---------------- phase 2: persistent recurrence, 16 WGs x 16 batch rows -----
#define SMEM_BYTES 191040
__global__ __launch_bounds__(256) void k_recur(
    const float* __restrict__ sx, const float* __restrict__ task,
    const float* __restrict__ hx0,
    const __bf16* __restrict__ wihp, const __bf16* __restrict__ whhp,
    const float* __restrict__ b_ih, const float* __restrict__ b_hh,
    const float* __restrict__ W_f,  const float* __restrict__ b_f,
    const float* __restrict__ W_u,  const float* __restrict__ b_u,
    const float* __restrict__ W_s,  const float* __restrict__ b_s,
    const float* __restrict__ W_pi, const float* __restrict__ b_pi,
    const float* __restrict__ W_beta, const float* __restrict__ b_beta,
    const unsigned* __restrict__ keys, float* __restrict__ out) {
  extern __shared__ char smem[];
  float*  sh_s   = (float*)smem;            // 16x256
  float*  sh_gi  = sh_s   + 4096;           // 16x768
  float*  sh_gh  = sh_gi  + 12288;          // 16x768
  float*  sh_h   = sh_gh  + 12288;          // 16x256
  float*  sh_M   = sh_h   + 4096;           // 16x16x32
  float*  sh_p   = sh_M   + 8192;           // 16x16
  float*  sh_pn  = sh_p   + 256;            // 16x16
  float*  sh_r   = sh_pn  + 256;            // 16x32
  float*  sh_rn  = sh_r   + 512;            // 16x32
  float*  sh_g   = sh_rn  + 512;            // 16x32
  float*  sh_cp  = sh_g   + 512;            // 16x16 partials for c
  float*  sh_lraw= sh_cp  + 256;            // 16x3 (padded 64)
  float*  sh_lg  = sh_lraw+ 64;             // 16x16 logits_g
  float*  sh_lbx = sh_lg  + 256;            // 16x2 x-part of logits_b
  float*  sh_bst = sh_lbx + 32;             // 16 boundary state b
  float*  sh_c   = sh_bst + 16;             // 16
  float*  sh_lp  = sh_c   + 16;             // 16 log-prob
  __bf16* sh_sb  = (__bf16*)(sh_lp + 16);   // 16x256 bf16 of s
  __bf16* sh_hb  = sh_sb + 4096;            // 16x256 bf16 of h

  const int tid = threadIdx.x, lane = tid & 31, wv = tid >> 5;
  const int b0 = blockIdx.x * 16;

  // ---- init state from hx0, M tile from task ----
  for (int i = tid; i < 16 * 512; i += 256) sh_M[i] = task[(size_t)b0 * 512 + i];
  for (int i = tid; i < 16 * 16; i += 256)
    sh_p[i] = hx0[(size_t)(b0 + i / 16) * TOT + (i % 16)];
  for (int i = tid; i < 16 * 32; i += 256) {
    int r_ = i / 32, cc = i % 32;
    sh_r[i] = hx0[(size_t)(b0 + r_) * TOT + 16 + cc];
    sh_g[i] = hx0[(size_t)(b0 + r_) * TOT + 304 + cc];
  }
  for (int i = tid; i < 16 * 256; i += 256) {
    int r_ = i >> 8, cc = i & 255;
    float hv = hx0[(size_t)(b0 + r_) * TOT + 48 + cc];
    sh_h[i] = hv; sh_hb[i] = (__bf16)hv;
  }
  if (tid < 16) sh_bst[tid] = hx0[(size_t)(b0 + tid) * TOT + 336];
  __syncthreads();

  for (int t = 0; t < TT; ++t) {
    // ---- B0: s = SX + [r,g,b] @ W_f_rgb^T + b_f ; capture lbx ----
    {
      int row = tid >> 4, gb = b0 + row;
      const float* sxrow = sx + ((size_t)t * BB + gb) * NPAD;
      float bstate = sh_bst[row];
      int c0 = (tid & 15) * 16;
      for (int cc = c0; cc < c0 + 16; ++cc) {
        const float* wf = W_f + (size_t)cc * 4161 + 4096;
        float a = sxrow[cc] + b_f[cc] + bstate * wf[64];
        for (int j = 0; j < 32; j++)
          a += sh_r[row * 32 + j] * wf[j] + sh_g[row * 32 + j] * wf[32 + j];
        sh_s[row * 256 + cc] = a;
        sh_sb[row * 256 + cc] = (__bf16)a;
      }
      if (tid < 32) {
        int r_ = tid >> 1, j = tid & 1;
        sh_lbx[tid] = sx[((size_t)t * BB + b0 + r_) * NPAD + 256 + j];
      }
    }
    __syncthreads();

    // ---- B1: GRU gates via WMMA (uses old h) ----
    gru_gemm(sh_sb, wihp, sh_gi, lane, wv);
    gru_gemm(sh_hb, whhp, sh_gh, lane, wv);

    // ---- B2: partials for c = sigmoid([s,h] @ W_u^T); l logits (old h) ----
    {
      int row = tid >> 4, seg = tid & 15;
      float a = 0.0f;
      for (int j = seg * 32; j < seg * 32 + 32; j++)
        a += (j < 256) ? sh_s[row * 256 + j] * W_u[j]
                       : sh_h[row * 256 + (j - 256)] * W_u[j];
      sh_cp[tid] = a;
      if (tid < 48) {
        int r_ = tid / 3, q = tid % 3;
        float a2 = b_s[q];
        for (int j = 0; j < 256; j++) a2 += sh_h[r_ * 256 + j] * W_s[q * 256 + j];
        sh_lraw[r_ * 3 + q] = a2;
      }
    }
    __syncthreads();

    // ---- B3: finalize c, softmax l, p_new ----
    if (tid < 16) {
      float ssum = b_u[0];
      for (int j = 0; j < 16; j++) ssum += sh_cp[tid * 16 + j];
      float c = sigmoidf_(ssum);
      sh_c[tid] = c;
      float l0 = sh_lraw[tid * 3], l1 = sh_lraw[tid * 3 + 1], l2 = sh_lraw[tid * 3 + 2];
      float mx = fmaxf(l0, fmaxf(l1, l2));
      float e0 = __expf(l0 - mx), e1 = __expf(l1 - mx), e2 = __expf(l2 - mx);
      float inv = 1.0f / (e0 + e1 + e2);
      e0 *= inv; e1 *= inv; e2 *= inv;
      for (int n = 0; n < 16; n++) {
        float pm1 = (n > 0)  ? sh_p[tid * 16 + n - 1] : 0.0f;
        float pp1 = (n < 15) ? sh_p[tid * 16 + n + 1] : 0.0f;
        sh_pn[tid * 16 + n] = e0 * pm1 + e1 * sh_p[tid * 16 + n] + e2 * pp1;
      }
    }
    __syncthreads();

    // ---- B4: r_new = p_new @ M ; h = c*h_new + (1-c)*h (GRU elementwise) ----
    for (int i = tid; i < 512; i += 256) {
      int row = i >> 5, sc = i & 31;
      float a = 0.0f;
      for (int n = 0; n < 16; n++) a += sh_pn[row * 16 + n] * sh_M[row * 512 + n * 32 + sc];
      sh_rn[i] = a;
    }
    {
      int row = tid >> 4;
      float c = sh_c[row];
      int c0 = (tid & 15) * 16;
      for (int cc = c0; cc < c0 + 16; ++cc) {
        float ir  = sh_gi[row * 768 + cc]       + b_ih[cc];
        float iz  = sh_gi[row * 768 + 256 + cc] + b_ih[256 + cc];
        float inn = sh_gi[row * 768 + 512 + cc] + b_ih[512 + cc];
        float hr  = sh_gh[row * 768 + cc]       + b_hh[cc];
        float hz  = sh_gh[row * 768 + 256 + cc] + b_hh[256 + cc];
        float hn  = sh_gh[row * 768 + 512 + cc] + b_hh[512 + cc];
        float rr = sigmoidf_(ir + hr);
        float z  = sigmoidf_(iz + hz);
        float nn = tanhf(inn + rr * hn);
        float hold = sh_h[row * 256 + cc];
        float hnew = (1.0f - z) * nn + z * hold;
        float hq = c * hnew + (1.0f - c) * hold;
        sh_h[row * 256 + cc] = hq;
        sh_hb[row * 256 + cc] = (__bf16)hq;
      }
    }
    __syncthreads();

    // ---- B5: gate p and r ----
    for (int i = tid; i < 512; i += 256) {
      int row = i >> 5;
      float c = sh_c[row];
      sh_r[i] = c * sh_rn[i] + (1.0f - c) * sh_r[i];
    }
    {
      int row = tid >> 4;
      float c = sh_c[row];
      sh_p[tid] = c * sh_pn[tid] + (1.0f - c) * sh_p[tid];
    }
    __syncthreads();

    // ---- B6: logits_g = [h,r] @ W_pi^T (updated h, r) ----
    {
      int row = tid >> 4, n = tid & 15;
      const float* wp_ = W_pi + (size_t)n * 288;
      float a = b_pi[n];
      for (int j = 0; j < 256; j++) a += sh_h[row * 256 + j] * wp_[j];
      for (int j = 0; j < 32; j++)  a += sh_r[row * 32 + j] * wp_[256 + j];
      sh_lg[tid] = a;
    }
    __syncthreads();

    // ---- B7: sample g and b (threefry gumbel-argmax), update g, b, logprob ----
    if (tid < 16) {
      int row = tid, gb = b0 + row;
      uint2 kg = make_uint2(keys[4 * t],     keys[4 * t + 1]);
      uint2 kb = make_uint2(keys[4 * t + 2], keys[4 * t + 3]);
      float lgv[16], mx = -1e30f;
      for (int n = 0; n < 16; n++) { lgv[n] = sh_lg[row * 16 + n]; mx = fmaxf(mx, lgv[n]); }
      float sum = 0.0f;
      for (int n = 0; n < 16; n++) sum += __expf(lgv[n] - mx);
      float lse = mx + __logf(sum);
      int gidx = 0; float best = -1e30f;
      for (int n = 0; n < 16; n++) {
        unsigned idx = (unsigned)(gb * 16 + n);
        float u = tf_uniform(tf_bits(kg, idx, 2048u));
        float v = lgv[n] - __logf(-__logf(u));
        if (v > best) { best = v; gidx = n; }
      }
      float lpg = lgv[gidx] - lse;
      float c = sh_c[row];
      for (int sc = 0; sc < 32; sc++) {
        float gv = sh_M[row * 512 + gidx * 32 + sc];
        sh_g[row * 32 + sc] = c * gv + (1.0f - c) * sh_g[row * 32 + sc];
      }
      float lb0[2];
      for (int j = 0; j < 2; j++) {
        const float* wb_ = W_beta + (size_t)j * 4128 + 4096;
        float a = sh_lbx[row * 2 + j] + b_beta[j];
        for (int sc = 0; sc < 32; sc++) a += sh_g[row * 32 + sc] * wb_[sc];
        lb0[j] = a;
      }
      float mb = fmaxf(lb0[0], lb0[1]);
      float lseb = mb + __logf(__expf(lb0[0] - mb) + __expf(lb0[1] - mb));
      int bidx = 0; float bbest = -1e30f;
      for (int j = 0; j < 2; j++) {
        unsigned idx = (unsigned)(gb * 2 + j);
        float u = tf_uniform(tf_bits(kb, idx, 256u));
        float v = lb0[j] - __logf(-__logf(u));
        if (v > bbest) { bbest = v; bidx = j; }
      }
      sh_bst[row] = (float)bidx;
      sh_lp[row]  = lpg + (lb0[bidx] - lseb);
    }
    __syncthreads();

    // ---- B8: write outputs for this timestep ----
    for (int i = tid; i < 16 * TOT; i += 256) {
      int row = i / TOT, o = i % TOT;
      int gb = b0 + row;
      float v;
      if      (o < 16)  v = sh_p[row * 16 + o];
      else if (o < 48)  v = sh_r[row * 32 + (o - 16)];
      else if (o < 304) v = sh_h[row * 256 + (o - 48)];
      else if (o < 336) v = sh_g[row * 32 + (o - 304)];
      else if (o == 336) v = sh_bst[row];
      else               v = sh_lp[row];
      out[((size_t)t * BB + gb) * TOT + o] = v;
    }
    __syncthreads();
  }
}

// ---------------- launch ----------------
extern "C" void kernel_launch(void* const* d_in, const int* in_sizes, int n_in,
                              void* d_out, int out_size, void* d_ws, size_t ws_size,
                              hipStream_t stream) {
  const float* obs    = (const float*)d_in[0];
  const float* task   = (const float*)d_in[1];
  const float* hx0    = (const float*)d_in[2];
  const float* W_ih   = (const float*)d_in[3];
  const float* W_hh   = (const float*)d_in[4];
  const float* b_ih   = (const float*)d_in[5];
  const float* b_hh   = (const float*)d_in[6];
  const float* W_f    = (const float*)d_in[7];
  const float* b_f    = (const float*)d_in[8];
  const float* W_u    = (const float*)d_in[9];
  const float* b_u    = (const float*)d_in[10];
  const float* W_s    = (const float*)d_in[11];
  const float* b_s    = (const float*)d_in[12];
  const float* W_pi   = (const float*)d_in[13];
  const float* b_pi   = (const float*)d_in[14];
  const float* W_beta = (const float*)d_in[15];
  const float* b_beta = (const float*)d_in[16];

  char* ws = (char*)d_ws;
  size_t off = 0;
  float* SX = (float*)(ws + off);            off += (size_t)TT * BB * NPAD * 4;    // 35.7 MB
  __bf16* WP  = (__bf16*)(ws + off);         off += (size_t)128 * NT1 * 512 * 2;   // 2.2 MB
  __bf16* WIH = (__bf16*)(ws + off);         off += (size_t)48 * 8 * 512 * 2;      // 384 KB
  __bf16* WHH = (__bf16*)(ws + off);         off += (size_t)48 * 8 * 512 * 2;      // 384 KB
  unsigned* KEYS = (unsigned*)(ws + off);    off += 512 * 4;

  k_keys   <<<1,    256, 0, stream>>>(KEYS);
  k_pack_wf<<<4352, 256, 0, stream>>>(W_f, W_beta, WP);
  k_pack_w <<<768,  256, 0, stream>>>(W_ih, WIH, 768, 256);
  k_pack_w <<<768,  256, 0, stream>>>(W_hh, WHH, 768, 256);
  k_sx_gemm<<<(TT * BB) / 128, 256, 0, stream>>>(obs, WP, SX);
  k_recur  <<<BB / 16, 256, SMEM_BYTES, stream>>>(
      SX, task, hx0, WIH, WHH, b_ih, b_hh, W_f, b_f, W_u, b_u,
      W_s, b_s, W_pi, b_pi, W_beta, b_beta, KEYS, (float*)d_out);
}
